// Conv1d_NN_48945447305766
// MI455X (gfx1250) — matrix operand order repaired and checked by reading, not compiled, not run on hardware
//
#include <hip/hip_runtime.h>
#include <hip/hip_bf16.h>
#include <stdint.h>

#define B_   16
#define C_   64
#define N_   2048
#define O_   64
#define KNN  3
#define KC   (C_ * KNN)   // 192, flattened (k,c) contraction axis

typedef __attribute__((ext_vector_type(16))) __bf16 v16bf;
typedef __attribute__((ext_vector_type(8)))  float  v8f;
typedef __attribute__((ext_vector_type(4)))  float  v4f;
typedef __attribute__((ext_vector_type(2)))  float  v2f;

union BfFrag { v16bf v; uint4 u[2]; };

// ---------------------------------------------------------------------------
// Prep: xt[b][n][c] = bf16(x[b][c][n]) (row-major per point, 128B rows),
//       norms[b][n] = sum_c x^2 (f32).
// ---------------------------------------------------------------------------
__global__ void prep_x(const float* __restrict__ x,
                       unsigned short* __restrict__ xt,
                       float* __restrict__ norms) {
  int t = blockIdx.x * blockDim.x + threadIdx.x;    // t = b*N + n
  if (t >= B_ * N_) return;
  int b = t / N_, n = t % N_;
  const float* xp = x + (size_t)b * C_ * N_ + n;
  unsigned short* op = xt + (size_t)t * C_;
  float s = 0.0f;
#pragma unroll
  for (int c = 0; c < C_; ++c) {
    float v = xp[(size_t)c * N_];
    s += v * v;
    union { float f; unsigned u; } cv; cv.f = v;
    unsigned u = cv.u;
    // round-to-nearest-even fp32 -> bf16
    unsigned short hb = (unsigned short)((u + 0x7FFFu + ((u >> 16) & 1u)) >> 16);
    op[c] = hb;
  }
  norms[t] = s;
}

// Pack W[o][c][k] -> Wf[o][k*64 + c] (f32), rows of the A matrix for the conv GEMM.
__global__ void prep_w(const float* __restrict__ W, float* __restrict__ Wf) {
  int t = blockIdx.x * blockDim.x + threadIdx.x;
  if (t >= O_ * C_ * KNN) return;
  int o   = t / (C_ * KNN);
  int rem = t % (C_ * KNN);
  int c   = rem / KNN;
  int k   = rem % KNN;
  Wf[(size_t)o * KC + k * C_ + c] = W[t];
}

// ---------------------------------------------------------------------------
// KNN: one wave per (batch, 16-query tile). Gram tiles via bf16 WMMA,
// fused running top-3 (value = |x_j|^2 - 2 x_i.x_j, same ordering as dist).
// Fully branchless selection so VALU co-executes with the XDL WMMA pipe.
// ---------------------------------------------------------------------------
__global__ void __launch_bounds__(256) knn_topk(const unsigned short* __restrict__ xt,
                                                const float* __restrict__ norms,
                                                int* __restrict__ idx) {
  int wave = blockIdx.x * (blockDim.x >> 5) + (threadIdx.x >> 5);
  int lane = threadIdx.x & 31;
  int b  = wave / (N_ / 16);
  int qt = wave % (N_ / 16);
  int h  = lane >> 4;       // half of the wave
  int ln = lane & 15;

  const unsigned short* xb = xt + (size_t)b * N_ * C_;
  const float* nb = norms + (size_t)b * N_;

  // B operand (queries), loaded once. 32x16 bf16 tile, K-window s:
  // lane holds channels 32*s + h*16 + [0..15]  (contiguous 32B).
  int q = qt * 16 + ln;
  v16bf bf0, bf1;
  {
    BfFrag t0, t1;
    const uint4* p0 = (const uint4*)(xb + (size_t)q * C_ + 0  + h * 16);
    const uint4* p1 = (const uint4*)(xb + (size_t)q * C_ + 32 + h * 16);
    t0.u[0] = p0[0]; t0.u[1] = p0[1];
    t1.u[0] = p1[0]; t1.u[1] = p1[1];
    bf0 = t0.v; bf1 = t1.v;
  }

  float d0 = __builtin_huge_valf(), d1 = d0, d2 = d0;
  int   i0 = 0, i1 = 0, i2 = 0;

#pragma unroll 2
  for (int jb = 0; jb < N_; jb += 16) {
    // A operand (candidates): lane's row = jb + ln; K chunk contiguous at +h*16 .. +h*16+15.
    const unsigned short* ap = xb + (size_t)(jb + ln) * C_;
    BfFrag ta, tb;
    ta.u[0] = *(const uint4*)(ap + h * 16);
    ta.u[1] = *(const uint4*)(ap + 8 + h * 16);
    tb.u[0] = *(const uint4*)(ap + 32 + h * 16);
    tb.u[1] = *(const uint4*)(ap + 40 + h * 16);

    // Norms for this lane's 8 candidate rows (contiguous, 32B aligned):
    // hoisted as two b128 loads so latency hides under the WMMAs.
    const float* np = nb + jb + 8 * h;
    v4f n0 = *(const v4f*)(np);
    v4f n1 = *(const v4f*)(np + 4);

    v8f acc = {};
    acc = __builtin_amdgcn_wmma_f32_16x16x32_bf16(false, ta.v, false, bf0, (short)0, acc, false, false);
    acc = __builtin_amdgcn_wmma_f32_16x16x32_bf16(false, tb.v, false, bf1, (short)0, acc, false, false);

    // C layout: VGPR r holds row M = r + 8h of column n = ln.
#pragma unroll
    for (int r = 0; r < 8; ++r) {
      float nrm = (r < 4) ? n0[r & 3] : n1[r & 3];
      float val = nrm - 2.0f * acc[r];
      int   j   = jb + 8 * h + r;
      // Branchless sorted-insert into (d0,i0) <= (d1,i1) <= (d2,i2).
      bool c0 = val < d0;
      bool c1 = val < d1;
      bool c2 = val < d2;
      float nd0 = c0 ? val : d0;
      int   ni0 = c0 ? j   : i0;
      float nd1 = c0 ? d0  : (c1 ? val : d1);
      int   ni1 = c0 ? i0  : (c1 ? j   : i1);
      float nd2 = c1 ? d1  : (c2 ? val : d2);
      int   ni2 = c1 ? i1  : (c2 ? j   : i2);
      d0 = nd0; i0 = ni0;
      d1 = nd1; i1 = ni1;
      d2 = nd2; i2 = ni2;
    }
  }

  // Merge the two halves' top-3 (lane L with lane L^16), stable on ties by index.
  float ad[4] = { d0, d1, d2, __builtin_huge_valf() };
  int   ai[4] = { i0, i1, i2, 0x7fffffff };
  float bd[4] = { __shfl_xor(d0, 16, 32), __shfl_xor(d1, 16, 32),
                  __shfl_xor(d2, 16, 32), __builtin_huge_valf() };
  int   bi[4] = { __shfl_xor(i0, 16, 32), __shfl_xor(i1, 16, 32),
                  __shfl_xor(i2, 16, 32), 0x7fffffff };
  int pa = 0, pb = 0, ri[3];
#pragma unroll
  for (int t = 0; t < 3; ++t) {
    bool takeA = (ad[pa] < bd[pb]) || (ad[pa] == bd[pb] && ai[pa] < bi[pb]);
    ri[t] = takeA ? ai[pa] : bi[pb];
    pa += takeA ? 1 : 0;
    pb += takeA ? 0 : 1;
  }
  if (h == 0) {
    int* op = idx + ((size_t)b * N_ + q) * KNN;
    op[0] = ri[0]; op[1] = ri[1]; op[2] = ri[2];
  }
}

// ---------------------------------------------------------------------------
// Conv: one wave per (batch, 16-point tile). out[o,n] = sum_{kappa} Wf[o,kappa] *
// x[c(kappa), idx[n, k(kappa)]] + bias[o], via f32 WMMA 16x16x4, 48 K-steps,
// 4 output row tiles (O=64).
// ---------------------------------------------------------------------------
__global__ void __launch_bounds__(256) conv_gemm(const float* __restrict__ x,
                                                 const float* __restrict__ Wf,
                                                 const float* __restrict__ bias,
                                                 const int* __restrict__ idx,
                                                 float* __restrict__ out) {
  int wave = blockIdx.x * (blockDim.x >> 5) + (threadIdx.x >> 5);
  int lane = threadIdx.x & 31;
  int b  = wave / (N_ / 16);
  int nt = wave % (N_ / 16);
  int h  = lane >> 4;
  int ln = lane & 15;

  int n = nt * 16 + ln;
  const int* ip = idx + ((size_t)b * N_ + n) * KNN;
  int j0 = ip[0], j1 = ip[1], j2 = ip[2];
  const float* xb = x + (size_t)b * C_ * N_;

  v8f acc0 = {}, acc1 = {}, acc2 = {}, acc3 = {};

#pragma unroll 4
  for (int kk = 0; kk < KC / 4; ++kk) {
    // B fragment: VGPR r -> K = r + 2h, column n = ln.
    v2f bf;
#pragma unroll
    for (int r = 0; r < 2; ++r) {
      int kappa = 4 * kk + r + 2 * h;
      int knn = kappa >> 6;            // which neighbor
      int c   = kappa & 63;            // which channel
      int j   = (knn == 0) ? j0 : ((knn == 1) ? j1 : j2);
      bf[r] = xb[(size_t)c * N_ + j];
    }
    // A fragments: lane row M = ln, K = {0,1} + 2h -> contiguous pair (8B aligned).
    v2f a0 = *(const v2f*)(Wf + (size_t)(0 * 16 + ln) * KC + 4 * kk + 2 * h);
    v2f a1 = *(const v2f*)(Wf + (size_t)(1 * 16 + ln) * KC + 4 * kk + 2 * h);
    v2f a2 = *(const v2f*)(Wf + (size_t)(2 * 16 + ln) * KC + 4 * kk + 2 * h);
    v2f a3 = *(const v2f*)(Wf + (size_t)(3 * 16 + ln) * KC + 4 * kk + 2 * h);
    acc0 = __builtin_amdgcn_wmma_f32_16x16x4_f32(false, a0, false, bf, (short)0, acc0, false, false);
    acc1 = __builtin_amdgcn_wmma_f32_16x16x4_f32(false, a1, false, bf, (short)0, acc1, false, false);
    acc2 = __builtin_amdgcn_wmma_f32_16x16x4_f32(false, a2, false, bf, (short)0, acc2, false, false);
    acc3 = __builtin_amdgcn_wmma_f32_16x16x4_f32(false, a3, false, bf, (short)0, acc3, false, false);
  }

  // Store: lane holds column n, rows M = r + 8h of each 16-row tile.
  float* ob = out + (size_t)b * O_ * N_ + n;
#pragma unroll
  for (int r = 0; r < 8; ++r) {
    int m = 8 * h + r;
    ob[(size_t)(0 * 16 + m) * N_] = acc0[r] + bias[0 * 16 + m];
    ob[(size_t)(1 * 16 + m) * N_] = acc1[r] + bias[1 * 16 + m];
    ob[(size_t)(2 * 16 + m) * N_] = acc2[r] + bias[2 * 16 + m];
    ob[(size_t)(3 * 16 + m) * N_] = acc3[r] + bias[3 * 16 + m];
  }
}

// ---------------------------------------------------------------------------
extern "C" void kernel_launch(void* const* d_in, const int* in_sizes, int n_in,
                              void* d_out, int out_size, void* d_ws, size_t ws_size,
                              hipStream_t stream) {
  const float* x    = (const float*)d_in[0];
  const float* W    = (const float*)d_in[1];
  const float* bias = (const float*)d_in[2];
  float* out = (float*)d_out;

  char* ws = (char*)d_ws;
  const size_t xt_bytes    = (size_t)B_ * N_ * C_ * 2;   // 4 MB bf16
  const size_t norms_bytes = (size_t)B_ * N_ * 4;        // 128 KB
  const size_t idx_bytes   = (size_t)B_ * N_ * KNN * 4;  // 384 KB
  unsigned short* xt    = (unsigned short*)(ws);
  float*          norms = (float*)(ws + xt_bytes);
  int*            idxb  = (int*)(ws + xt_bytes + norms_bytes);
  float*          Wf    = (float*)(ws + xt_bytes + norms_bytes + idx_bytes);

  prep_x<<<(B_ * N_ + 255) / 256, 256, 0, stream>>>(x, xt, norms);
  prep_w<<<(O_ * C_ * KNN + 255) / 256, 256, 0, stream>>>(W, Wf);

  // 16 batches * 128 query tiles = 2048 waves; 8 waves (256 thr) per block.
  knn_topk<<<(B_ * (N_ / 16)) / 8, 256, 0, stream>>>(xt, norms, idxb);
  conv_gemm<<<(B_ * (N_ / 16)) / 8, 256, 0, stream>>>(x, Wf, bias, idxb, out);
}
